// SelfAttention_15195594293305
// MI455X (gfx1250) — matrix operand order repaired and checked by reading, not compiled, not run on hardware
//
#include <hip/hip_runtime.h>

// ---------------------------------------------------------------------------
// MI455X (gfx1250) self-attention w/ MLP projections.
// ~1.8 TFLOP of GEMM vs <1 GB of traffic -> fully WMMA-bound design:
//   * all GEMMs on v_wmma_f32_16x16x32_bf16 (f32 accumulate)
//   * weights transpose-cast to bf16 so every GEMM is the coalesced NT case
//   * double-buffered LDS tiles filled by GLOBAL_LOAD_ASYNC_TO_LDS_B128
//     (ASYNCcnt) so global->LDS overlaps WMMA compute
//   * 64x64 wave tile (4x4 fragments): 16 WMMAs per 16 ds_load_b128
// ---------------------------------------------------------------------------

typedef __attribute__((ext_vector_type(16))) __bf16 v16bf;
typedef __attribute__((ext_vector_type(8)))  float  v8f;
typedef int v4i __attribute__((vector_size(16)));   // matches builtin's V4i

union FragBf { v16bf v; uint4 q[2]; };

#if defined(__AMDGCN__) && defined(__gfx1250__) && \
    __has_builtin(__builtin_amdgcn_global_load_async_to_lds_b128)
#define USE_ASYNC 1
#else
#define USE_ASYNC 0
#endif

__device__ __forceinline__ void cp16_g2l(const unsigned short* g, unsigned short* l) {
#if USE_ASYNC
  __builtin_amdgcn_global_load_async_to_lds_b128(
      (__attribute__((address_space(1))) v4i*)(g),
      (__attribute__((address_space(3))) v4i*)(l), 0, 0);
#else
  *(uint4*)l = *(const uint4*)g;
#endif
}

__device__ __forceinline__ void wait_async_all() {
#if USE_ASYNC
#if __has_builtin(__builtin_amdgcn_s_wait_asynccnt)
  __builtin_amdgcn_s_wait_asynccnt(0);
#else
  asm volatile("s_wait_asynccnt 0x0" ::: "memory");
#endif
#endif
}

__device__ __forceinline__ unsigned short f32_to_bf16_rne(float f) {
  unsigned int u = __float_as_uint(f);
  u += 0x7fffu + ((u >> 16) & 1u);   // round-to-nearest-even
  return (unsigned short)(u >> 16);
}

// ------------------------------ cast kernel --------------------------------
__global__ void cast_f32_to_bf16(const float* __restrict__ in,
                                 unsigned short* __restrict__ out,
                                 long long n) {
  long long i = (long long)blockIdx.x * blockDim.x + threadIdx.x;
  long long stride = (long long)gridDim.x * blockDim.x;
  for (; i < n; i += stride) out[i] = f32_to_bf16_rne(in[i]);
}

// ------------------- transpose-cast: f32 [R][C] -> bf16 [C][R] -------------
__global__ void transpose_cast_f32_bf16(const float* __restrict__ in,
                                        unsigned short* __restrict__ out,
                                        int R, int C) {
  __shared__ unsigned short t[32][33];
  const int c0 = blockIdx.x * 32, r0 = blockIdx.y * 32;
  const int tx = threadIdx.x, ty = threadIdx.y;
#pragma unroll
  for (int j = ty; j < 32; j += 8)
    t[j][tx] = f32_to_bf16_rne(in[(long long)(r0 + j) * C + (c0 + tx)]);
  __syncthreads();
#pragma unroll
  for (int j = ty; j < 32; j += 8)
    out[(long long)(c0 + j) * R + (r0 + tx)] = t[tx][j];
}

// ------------------- transpose: bf16 [R][C] -> bf16 [C][R], batched --------
__global__ void transpose_bf16(const unsigned short* __restrict__ in,
                               unsigned short* __restrict__ out,
                               int R, int C, long long sBat) {
  __shared__ unsigned short t[32][33];
  in  += (long long)blockIdx.z * sBat;
  out += (long long)blockIdx.z * sBat;
  const int c0 = blockIdx.x * 32, r0 = blockIdx.y * 32;
  const int tx = threadIdx.x, ty = threadIdx.y;
#pragma unroll
  for (int j = ty; j < 32; j += 8)
    t[j][tx] = in[(long long)(r0 + j) * C + (c0 + tx)];
  __syncthreads();
#pragma unroll
  for (int j = ty; j < 32; j += 8)
    out[(long long)(c0 + j) * R + (r0 + tx)] = t[tx][j];
}

// ------------------------------ WMMA GEMM (NT) -----------------------------
// C[M,N] = op(A[M,K] * Bt[N,K]^T + bias) ; A/Bt bf16 row-major, f32 acc.
// 256 threads = 8 wave32 ; block tile 128x256xBK ; wave tile 64x64 (4x4 frags)
// Double-buffered LDS, async global->LDS copies overlap WMMA compute.
#define BM 128
#define BN 256
#define BK 32
#define BKP 40   // +8 bf16 pad against LDS bank conflicts

template<bool RELU, bool OUT_BF16>
__global__ __launch_bounds__(256)
void gemm_nt_bf16_wmma(const unsigned short* __restrict__ A,   // [M,K]
                       const unsigned short* __restrict__ Bt,  // [N,K]
                       const float* __restrict__ bias,
                       void* __restrict__ Cout,
                       int N, int K,
                       long long sA, long long sB, long long sC) {
  __shared__ unsigned short As[2][BM][BKP];
  __shared__ unsigned short Bs[2][BN][BKP];

  const int tid  = threadIdx.x;
  const int lane = tid & 31;
  const int wave = tid >> 5;
  const int wm   = wave & 1;          // wave M offset = wm*64
  const int wn   = wave >> 1;         // wave N offset = wn*64
  const int bm   = blockIdx.y * BM;
  const int bn   = blockIdx.x * BN;

  A  += (long long)blockIdx.z * sA;
  Bt += (long long)blockIdx.z * sB;

  const int tr = lane & 15;           // row/col within 16x16 fragment
  const int hi = lane >> 4;           // lane half selector

  v8f acc[4][4];
#pragma unroll
  for (int i = 0; i < 4; ++i)
#pragma unroll
    for (int j = 0; j < 4; ++j)
#pragma unroll
      for (int r = 0; r < 8; ++r) acc[i][j][r] = 0.0f;

  auto load_tile = [&](int buf, int k0) {
#pragma unroll
    for (int it = 0; it < 2; ++it) {          // A: 128x32 bf16 = 512 x b128
      int id = tid + it * 256;
      int r = id >> 2, c = (id & 3) << 3;
      cp16_g2l(A + (long long)(bm + r) * K + (k0 + c), &As[buf][r][c]);
    }
#pragma unroll
    for (int it = 0; it < 4; ++it) {          // B: 256x32 bf16 = 1024 x b128
      int id = tid + it * 256;
      int r = id >> 2, c = (id & 3) << 3;
      cp16_g2l(Bt + (long long)(bn + r) * K + (k0 + c), &Bs[buf][r][c]);
    }
  };

  const int kTiles = K / BK;
  load_tile(0, 0);

  for (int kt = 0; kt < kTiles; ++kt) {
    const int cur = kt & 1;
    wait_async_all();        // this thread's copies for tile kt are in LDS
    __syncthreads();         // everyone's copies visible; prev buffer free
    if (kt + 1 < kTiles) load_tile(cur ^ 1, (kt + 1) * BK);  // overlap

    FragBf a[4], b[4];
    // A frag (16x32): lane half 0 -> K {0..7,16..23}, half 1 -> {8..15,24..31}
#pragma unroll
    for (int mi = 0; mi < 4; ++mi) {
      const unsigned short* p = &As[cur][wm * 64 + mi * 16 + tr][hi * 8];
      a[mi].q[0] = *(const uint4*)p;
      a[mi].q[1] = *(const uint4*)(p + 16);
    }
    // B frag (32x16): lane half 0 -> K {0..7,16..23}, half 1 -> {8..15,24..31}
#pragma unroll
    for (int ni = 0; ni < 4; ++ni) {
      const unsigned short* p = &Bs[cur][wn * 64 + ni * 16 + tr][hi * 8];
      b[ni].q[0] = *(const uint4*)p;
      b[ni].q[1] = *(const uint4*)(p + 16);
    }
#pragma unroll
    for (int mi = 0; mi < 4; ++mi)
#pragma unroll
      for (int ni = 0; ni < 4; ++ni)
        acc[mi][ni] = __builtin_amdgcn_wmma_f32_16x16x32_bf16(
            false, a[mi].v, false, b[ni].v, (short)0, acc[mi][ni],
            false, false);
  }

  // ---- epilogue: C frag lane layout: VGPR r -> M = r + hi*8, N = tr ----
  const long long cbase = (long long)blockIdx.z * sC;
#pragma unroll
  for (int mi = 0; mi < 4; ++mi) {
#pragma unroll
    for (int ni = 0; ni < 4; ++ni) {
      const int n = bn + wn * 64 + ni * 16 + tr;
      const float bv = bias ? bias[n] : 0.0f;
#pragma unroll
      for (int r = 0; r < 8; ++r) {
        const int m = bm + wm * 64 + mi * 16 + hi * 8 + r;
        float v = acc[mi][ni][r] + bv;
        if (RELU) v = fmaxf(v, 0.0f);
        if (OUT_BF16)
          ((unsigned short*)Cout)[cbase + (long long)m * N + n] = f32_to_bf16_rne(v);
        else
          ((float*)Cout)[cbase + (long long)m * N + n] = v;
      }
    }
  }
}

// ----------------------- softmax over axis=1 (rows) ------------------------
// probs[b,s,t] = exp(sc[b,s,t]) / sum_s exp(sc[b,s,t]).
// Block (32,8): 32 columns per block, 8-way split over s with LDS reduce.
__global__ void softmax_axis1(const float* __restrict__ scores,
                              unsigned short* __restrict__ probs, int S) {
  __shared__ float red[8][32];
  const int tx = threadIdx.x, ty = threadIdx.y;
  const int t = blockIdx.x * 32 + tx;
  const long long base = (long long)blockIdx.y * S * S;
  const float* sc = scores + base + t;
  unsigned short* pr = probs + base + t;
  const int chunk = S >> 3;
  const int s0 = ty * chunk, s1 = s0 + chunk;

  float mx = -3.402823466e+38f;
  for (int s = s0; s < s1; ++s) mx = fmaxf(mx, sc[(long long)s * S]);
  red[ty][tx] = mx;
  __syncthreads();
#pragma unroll
  for (int j = 0; j < 8; ++j) mx = fmaxf(mx, red[j][tx]);
  __syncthreads();

  float sum = 0.0f;
  for (int s = s0; s < s1; ++s) sum += __expf(sc[(long long)s * S] - mx);
  red[ty][tx] = sum;
  __syncthreads();
  sum = 0.0f;
#pragma unroll
  for (int j = 0; j < 8; ++j) sum += red[j][tx];
  const float inv = 1.0f / sum;
  for (int s = s0; s < s1; ++s)
    pr[(long long)s * S] = f32_to_bf16_rne(__expf(sc[(long long)s * S] - mx) * inv);
}

// ------------------------------- launcher ----------------------------------
extern "C" void kernel_launch(void* const* d_in, const int* in_sizes, int n_in,
                              void* d_out, int out_size, void* d_ws, size_t ws_size,
                              hipStream_t stream) {
  (void)in_sizes; (void)n_in; (void)out_size; (void)ws_size;
  const int D = 2048, H = 8192, B = 4, S = 2048;
  const long long BSD = (long long)B * S * D;   // 16,777,216 (per-QKV elems)
  const long long DH  = (long long)D * H;       // 16,777,216
  const long long BSH = (long long)B * S * H;   // 67,108,864
  const long long SS  = (long long)S * S;

  const float* x     = (const float*)d_in[0];
  const float* w1[3] = {(const float*)d_in[1], (const float*)d_in[5], (const float*)d_in[9]};
  const float* b1[3] = {(const float*)d_in[2], (const float*)d_in[6], (const float*)d_in[10]};
  const float* w2[3] = {(const float*)d_in[3], (const float*)d_in[7], (const float*)d_in[11]};
  const float* b2[3] = {(const float*)d_in[4], (const float*)d_in[8], (const float*)d_in[12]};

  // workspace layout (bf16 buffers), total ~402 MB
  char* ws = (char*)d_ws;
  unsigned short* xb   = (unsigned short*)(ws);
  unsigned short* w1t  = (unsigned short*)(ws + 2 * BSD);                    // [H][D]
  unsigned short* w2t  = (unsigned short*)(ws + 2 * BSD + 2 * DH);           // [D][H]
  unsigned short* hid  = (unsigned short*)(ws + 2 * BSD + 4 * DH);           // [B*S][H]
  unsigned short* Qb   = (unsigned short*)(ws + 2 * BSD + 4 * DH + 2 * BSH);
  unsigned short* Kb   = Qb + BSD;
  unsigned short* Vb   = Kb + BSD;
  unsigned short* Pb   = Vb + BSD;       // probs  [b][s][t]
  unsigned short* Pt   = Pb + B * SS;    // probsT [b][t][s]
  float* scores = (float*)d_out;         // d_out doubles as f32 scores scratch
  unsigned short* qkv[3] = {Qb, Kb, Vb};

  const int MB = B * S;  // 8192 rows for the MLP GEMMs

  cast_f32_to_bf16<<<4096, 256, 0, stream>>>(x, xb, BSD);

  for (int p = 0; p < 3; ++p) {
    // w1 [D][H] -> w1t [H][D] bf16
    transpose_cast_f32_bf16<<<dim3(H / 32, D / 32), dim3(32, 8), 0, stream>>>(
        w1[p], w1t, D, H);
    // hid = relu(xb @ w1 + b1)   [8192 x 8192], K = D
    gemm_nt_bf16_wmma<true, true>
        <<<dim3(H / BN, MB / BM, 1), 256, 0, stream>>>(
            xb, w1t, b1[p], hid, H, D, 0, 0, 0);
    // w2 [H][D] -> w2t [D][H] bf16
    transpose_cast_f32_bf16<<<dim3(D / 32, H / 32), dim3(32, 8), 0, stream>>>(
        w2[p], w2t, H, D);
    // qkv = hid @ w2 + b2        [8192 x 2048], K = H
    gemm_nt_bf16_wmma<false, true>
        <<<dim3(D / BN, MB / BM, 1), 256, 0, stream>>>(
            hid, w2t, b2[p], qkv[p], D, H, 0, 0, 0);
  }

  // scores[b] = Q[b] @ K[b]^T    [2048 x 2048], K = D (Kb rows are coalesced)
  gemm_nt_bf16_wmma<false, false>
      <<<dim3(S / BN, S / BM, B), 256, 0, stream>>>(
          Qb, Kb, nullptr, scores, S, D,
          (long long)S * D, (long long)S * D, SS);

  // probs = softmax over axis 1 (column softmax of the SxS matrix), bf16
  softmax_axis1<<<dim3(S / 32, B), dim3(32, 8), 0, stream>>>(scores, Pb, S);

  // probsT[b][t][s] = probs[b][s][t]
  transpose_bf16<<<dim3(S / 32, S / 32, B), dim3(32, 8), 0, stream>>>(
      Pb, Pt, S, S, SS);

  // out[b] = V[b] @ probs[b]     [2048 x 2048], K = D(=S), NT via probsT
  gemm_nt_bf16_wmma<false, false>
      <<<dim3(S / BN, S / BM, B), 256, 0, stream>>>(
          Vb, Pt, nullptr, (float*)d_out, S, S,
          (long long)S * D, SS, SS);
}
